// Attention_45595372814689
// MI455X (gfx1250) — matrix operand order repaired and checked by reading, not compiled, not run on hardware
//
#include <hip/hip_runtime.h>
#include <hip/hip_bf16.h>

typedef __attribute__((ext_vector_type(16))) _Float16 v16h;
typedef __attribute__((ext_vector_type(8)))  _Float16 v8h;
typedef __attribute__((ext_vector_type(4)))  _Float16 v4h;
typedef __attribute__((ext_vector_type(8)))  float    v8f;

#define BQ     2
#define NSEQ   2048
#define DMODEL 1024
#define NH     16
#define HD     64

// ---------------------------------------------------------------------------
// WMMA fragment loaders from LDS (f16, row-major [row][ldm]), per ISA 7.12.2.
// A 16x32 f16: lane m=lane&15, khalf=lane>>4; elems 0..7 = K khalf*8+0..7,
//              elems 8..15 = K 16+khalf*8+0..7  (two 16B contiguous runs).
// B 32x16 f16: lane n=lane&15, kgroup=lane>>4; elems 0..7 = K kgroup*8+0..7,
//              elems 8..15 = K 16+kgroup*8+0..7 (same as A).
// ---------------------------------------------------------------------------
__device__ inline v16h load_frag_a(const _Float16* s, int row0, int kbase, int ldm) {
    int lane = threadIdx.x & 31;
    int m = lane & 15, kh = lane >> 4;
    const _Float16* p0 = s + (row0 + m) * ldm + kbase + kh * 8;
    v8h lo = *(const v8h*)p0;
    v8h hi = *(const v8h*)(p0 + 16);
    v16h a;
#pragma unroll
    for (int i = 0; i < 8; ++i) { a[i] = lo[i]; a[i + 8] = hi[i]; }
    return a;
}

__device__ inline v16h load_frag_b(const _Float16* s, int col0, int kbase, int ldm) {
    int lane = threadIdx.x & 31;
    int n = lane & 15, kg = lane >> 4;
    const _Float16* p0 = s + (col0 + n) * ldm + kbase + kg * 8;
    v8h lo = *(const v8h*)p0;
    v8h hi = *(const v8h*)(p0 + 16);
    v16h b;
#pragma unroll
    for (int i = 0; i < 8; ++i) { b[i] = lo[i]; b[i + 8] = hi[i]; }
    return b;
}

__device__ inline v8f wmma_f16(v16h a, v16h b, v8f c) {
    return __builtin_amdgcn_wmma_f32_16x16x32_f16(false, a, false, b,
                                                  (short)0, c, false, false);
}

// ---------------------------------------------------------------------------
// fp32 -> f16 conversion (vectorized, n must be a multiple of 4)
// ---------------------------------------------------------------------------
__global__ __launch_bounds__(256)
void f32_to_f16_kernel(const float* __restrict__ src, _Float16* __restrict__ dst,
                       int n4) {
    int i = blockIdx.x * 256 + threadIdx.x;
    if (i < n4) {
        float4 v = ((const float4*)src)[i];
        v4h h = { (_Float16)v.x, (_Float16)v.y, (_Float16)v.z, (_Float16)v.w };
        ((v4h*)dst)[i] = h;
    }
}

// ---------------------------------------------------------------------------
// GEMM: C[M,N] = A[M,K](f16) * W[N,K](f16)^T + bias[N]  -> fp32 C
// Block 256 thr = 8 waves, tile 256(M) x 128(N), K-step 32, double-buffered
// LDS with one barrier/step. Wave grid 4(M) x 2(N): each wave 64x64
// (4x4 WMMA tiles -> 16 wmma per 16 ds_load_b128 per K-step).
// ---------------------------------------------------------------------------
__global__ __launch_bounds__(256)
void gemm_wmma_kernel(const _Float16* __restrict__ A, const _Float16* __restrict__ W,
                      const float* __restrict__ bias, float* __restrict__ C,
                      int M, int Nout, int K) {
    const int LD = 40;                        // f16 row stride (80B, 16B aligned)
    __shared__ __align__(16) _Float16 As[2][256 * 40];   // 40 KB
    __shared__ __align__(16) _Float16 Bs[2][128 * 40];   // 20 KB

    const int jBase = blockIdx.x * 128;
    const int mBase = blockIdx.y * 256;
    const int tid  = threadIdx.x;
    const int wave = tid >> 5, lane = tid & 31;
    const int wm = wave >> 1, wn = wave & 1;   // 4 x 2 wave grid

    v8f acc[4][4];
#pragma unroll
    for (int i = 0; i < 4; ++i)
#pragma unroll
        for (int j = 0; j < 4; ++j) acc[i][j] = (v8f){0.f,0.f,0.f,0.f,0.f,0.f,0.f,0.f};

    // Per-thread staging: A tile 256x32 halves = 1024 8-half chunks (4/thread),
    // B tile 128x32 halves = 512 chunks (2/thread).
    v8h pa[4], pb[2];
    const int ar[4] = { (tid + 0*256) >> 2, (tid + 1*256) >> 2,
                        (tid + 2*256) >> 2, (tid + 3*256) >> 2 };
    const int ac = (tid & 3) * 8;
    const int br[2] = { (tid + 0*256) >> 2, (tid + 1*256) >> 2 };

    const int KT = K / 32;
    // Prime buffer 0
#pragma unroll
    for (int i = 0; i < 4; ++i)
        pa[i] = *(const v8h*)(A + (size_t)(mBase + ar[i]) * K + ac);
#pragma unroll
    for (int i = 0; i < 2; ++i)
        pb[i] = *(const v8h*)(W + (size_t)(jBase + br[i]) * K + ac);
#pragma unroll
    for (int i = 0; i < 4; ++i) *(v8h*)(&As[0][ar[i] * LD + ac]) = pa[i];
#pragma unroll
    for (int i = 0; i < 2; ++i) *(v8h*)(&Bs[0][br[i] * LD + ac]) = pb[i];
    __syncthreads();

    for (int kt = 0; kt < KT; ++kt) {
        const int cur = kt & 1, nxt = cur ^ 1;
        if (kt + 1 < KT) {
            int k0 = (kt + 1) * 32;
#pragma unroll
            for (int i = 0; i < 4; ++i)
                pa[i] = *(const v8h*)(A + (size_t)(mBase + ar[i]) * K + k0 + ac);
#pragma unroll
            for (int i = 0; i < 2; ++i)
                pb[i] = *(const v8h*)(W + (size_t)(jBase + br[i]) * K + k0 + ac);
        }

        v16h bf[4];
#pragma unroll
        for (int ni = 0; ni < 4; ++ni)
            bf[ni] = load_frag_b(Bs[cur], wn * 64 + ni * 16, 0, LD);
#pragma unroll
        for (int mi = 0; mi < 4; ++mi) {
            v16h af = load_frag_a(As[cur], wm * 64 + mi * 16, 0, LD);
#pragma unroll
            for (int ni = 0; ni < 4; ++ni)
                acc[mi][ni] = wmma_f16(af, bf[ni], acc[mi][ni]);
        }

        if (kt + 1 < KT) {
#pragma unroll
            for (int i = 0; i < 4; ++i) *(v8h*)(&As[nxt][ar[i] * LD + ac]) = pa[i];
#pragma unroll
            for (int i = 0; i < 2; ++i) *(v8h*)(&Bs[nxt][br[i] * LD + ac]) = pb[i];
        }
        __syncthreads();
    }

    // Epilogue: C layout lane n=lane&15, hi=lane>>4; VGPR r -> row r+8*hi.
    const int n = lane & 15, hi = lane >> 4;
#pragma unroll
    for (int mi = 0; mi < 4; ++mi)
#pragma unroll
        for (int ni = 0; ni < 4; ++ni) {
            int col = jBase + wn * 64 + ni * 16 + n;
            float bv = bias[col];
#pragma unroll
            for (int r = 0; r < 8; ++r) {
                int row = mBase + wm * 64 + mi * 16 + r + 8 * hi;
                C[(size_t)row * Nout + col] = acc[mi][ni][r] + bv;
            }
        }
}

// ---------------------------------------------------------------------------
// RoPE on q,k + pack q,k,v to f16 in [b,h,n,64]. One thread per (b,h,pos,d/2).
// ---------------------------------------------------------------------------
__global__ __launch_bounds__(256)
void rope_pack_kernel(const float* __restrict__ qkv, const float* __restrict__ rope,
                      _Float16* __restrict__ qh, _Float16* __restrict__ kh,
                      _Float16* __restrict__ vh) {
    int idx = blockIdx.x * 256 + threadIdx.x;   // B*H*N*32 total
    int d2  = idx & 31; int t = idx >> 5;
    int pos = t & (NSEQ - 1); t >>= 11;
    int h   = t & (NH - 1);   int b = t >> 4;

    size_t mrow = ((size_t)(b * NSEQ + pos)) * (3 * DMODEL);
    int c = h * HD + 2 * d2;
    float q0 = qkv[mrow + c],               q1 = qkv[mrow + c + 1];
    float k0 = qkv[mrow + DMODEL + c],      k1 = qkv[mrow + DMODEL + c + 1];
    float v0 = qkv[mrow + 2 * DMODEL + c],  v1 = qkv[mrow + 2 * DMODEL + c + 1];
    float cs = rope[pos * HD + 2 * d2], sn = rope[pos * HD + 2 * d2 + 1];

    size_t o = (((size_t)(b * NH + h)) * NSEQ + pos) * HD + 2 * d2;
    qh[o]     = (_Float16)(q0 * cs - q1 * sn);
    qh[o + 1] = (_Float16)(q0 * sn + q1 * cs);
    kh[o]     = (_Float16)(k0 * cs - k1 * sn);
    kh[o + 1] = (_Float16)(k0 * sn + k1 * cs);
    vh[o]     = (_Float16)v0;
    vh[o + 1] = (_Float16)v1;
}

// ---------------------------------------------------------------------------
// Flash attention: one block per (b,h,q-tile of 64). 128 thr = 4 waves, each
// wave owns 16 q rows. KV tiles of 64. S and O via WMMA, softmax fp32 online.
// Output f16 [b, n, h*64+d] (A operand of the projection GEMM).
// ---------------------------------------------------------------------------
__global__ __launch_bounds__(128)
void flash_attn_kernel(const _Float16* __restrict__ qh, const _Float16* __restrict__ kh,
                       const _Float16* __restrict__ vh, _Float16* __restrict__ oattn) {
    const int LD = 72;                       // 144B rows, 16B aligned
    __shared__ __align__(16) _Float16 Qs[64 * 72];
    __shared__ __align__(16) _Float16 Ks[64 * 72];
    __shared__ __align__(16) _Float16 Vs[64 * 72];   // transposed: Vs[d][kk]
    __shared__ __align__(16) _Float16 Ps[4][16 * 72];

    const int bid = blockIdx.x;
    const int qt = bid & 31, h = (bid >> 5) & 15, b = bid >> 9;
    const int tid = threadIdx.x, wave = tid >> 5, lane = tid & 31;
    const int n = lane & 15, hi = lane >> 4;
    const size_t headBase = ((size_t)(b * NH + h)) * NSEQ * HD;
    const float sc = 0.125f;                 // 1/sqrt(64)

    // Load Q tile: 64x64 f16 = 1024 chunks of 4 halves, 8 per thread.
#pragma unroll
    for (int i = 0; i < 8; ++i) {
        int cidx = tid + i * 128;
        int r = cidx >> 4, c4 = (cidx & 15) * 4;
        *(v4h*)(Qs + r * LD + c4) =
            *(const v4h*)(qh + headBase + (size_t)(qt * 64 + r) * HD + c4);
    }

    float m_i[8], l_i[8];
    v8f o_acc[4];
#pragma unroll
    for (int r = 0; r < 8; ++r) { m_i[r] = -1e30f; l_i[r] = 0.f; }
#pragma unroll
    for (int dt = 0; dt < 4; ++dt) o_acc[dt] = (v8f){0.f,0.f,0.f,0.f,0.f,0.f,0.f,0.f};

    for (int kt = 0; kt < NSEQ / 64; ++kt) {
        __syncthreads();
        // Stage K (as-is) and V (transposed) tiles.
#pragma unroll
        for (int i = 0; i < 8; ++i) {
            int cidx = tid + i * 128;
            int r = cidx >> 4, c4 = (cidx & 15) * 4;
            *(v4h*)(Ks + r * LD + c4) =
                *(const v4h*)(kh + headBase + (size_t)(kt * 64 + r) * HD + c4);
            v4h vv = *(const v4h*)(vh + headBase + (size_t)(kt * 64 + r) * HD + c4);
#pragma unroll
            for (int j = 0; j < 4; ++j) Vs[(c4 + j) * LD + r] = vv[j];
        }
        __syncthreads();

        // S = Q * K^T  (wave's 16 q-rows x 64 kv-cols)
        v8f s_acc[4];
#pragma unroll
        for (int nt = 0; nt < 4; ++nt) s_acc[nt] = (v8f){0.f,0.f,0.f,0.f,0.f,0.f,0.f,0.f};
#pragma unroll
        for (int ks = 0; ks < 2; ++ks) {
            v16h a = load_frag_a(Qs, wave * 16, ks * 32, LD);
#pragma unroll
            for (int nt = 0; nt < 4; ++nt) {
                v16h bf = load_frag_b(Ks, nt * 16, ks * 32, LD);
                s_acc[nt] = wmma_f16(a, bf, s_acc[nt]);
            }
        }

        // Online softmax per row (row = r + 8*hi of the wave's 16 rows).
#pragma unroll
        for (int r = 0; r < 8; ++r) {
            float mx = -1e30f;
#pragma unroll
            for (int nt = 0; nt < 4; ++nt) mx = fmaxf(mx, s_acc[nt][r]);
            mx *= sc;
#pragma unroll
            for (int off = 1; off < 16; off <<= 1)
                mx = fmaxf(mx, __shfl_xor(mx, off, 16));
            float mnew = fmaxf(m_i[r], mx);
            float corr = __expf(m_i[r] - mnew);
            float rsum = 0.f;
#pragma unroll
            for (int nt = 0; nt < 4; ++nt) {
                float p = __expf(s_acc[nt][r] * sc - mnew);
                rsum += p;
                Ps[wave][(r + 8 * hi) * LD + nt * 16 + n] = (_Float16)p;
            }
#pragma unroll
            for (int off = 1; off < 16; off <<= 1)
                rsum += __shfl_xor(rsum, off, 16);
            l_i[r] = l_i[r] * corr + rsum;
            m_i[r] = mnew;
#pragma unroll
            for (int dt = 0; dt < 4; ++dt) o_acc[dt][r] *= corr;
        }

        // O += P * V  (Ps is per-wave; same-wave LDS RAW handled by s_wait_dscnt)
#pragma unroll
        for (int ks = 0; ks < 2; ++ks) {
            v16h a = load_frag_a(&Ps[wave][0], 0, ks * 32, LD);
#pragma unroll
            for (int dt = 0; dt < 4; ++dt) {
                v16h bf = load_frag_b(Vs, dt * 16, ks * 32, LD);
                o_acc[dt] = wmma_f16(a, bf, o_acc[dt]);
            }
        }
    }

    // Normalize and write f16 [b, row, h*64 + d].
#pragma unroll
    for (int dt = 0; dt < 4; ++dt)
#pragma unroll
        for (int r = 0; r < 8; ++r) {
            int row = qt * 64 + wave * 16 + r + 8 * hi;
            oattn[((size_t)(b * NSEQ + row)) * DMODEL + h * HD + dt * 16 + n] =
                (_Float16)(o_acc[dt][r] * (1.0f / l_i[r]));
        }
}

// ---------------------------------------------------------------------------
extern "C" void kernel_launch(void* const* d_in, const int* in_sizes, int n_in,
                              void* d_out, int out_size, void* d_ws, size_t ws_size,
                              hipStream_t stream) {
    const float* x      = (const float*)d_in[0];
    const float* rope   = (const float*)d_in[1];
    const float* qkv_w  = (const float*)d_in[2];
    const float* qkv_b  = (const float*)d_in[3];
    const float* proj_w = (const float*)d_in[4];
    const float* proj_b = (const float*)d_in[5];
    float* out = (float*)d_out;

    const int M = BQ * NSEQ;                       // 4096
    char* ws = (char*)d_ws;
    size_t off = 0;
    float*    qkv   = (float*)(ws + off); off += (size_t)M * 3 * DMODEL * sizeof(float);
    _Float16* qh    = (_Float16*)(ws + off); off += (size_t)BQ * NH * NSEQ * HD * 2;
    _Float16* kh    = (_Float16*)(ws + off); off += (size_t)BQ * NH * NSEQ * HD * 2;
    _Float16* vh    = (_Float16*)(ws + off); off += (size_t)BQ * NH * NSEQ * HD * 2;
    _Float16* xh    = (_Float16*)(ws + off); off += (size_t)M * DMODEL * 2;
    _Float16* qwh   = (_Float16*)(ws + off); off += (size_t)3 * DMODEL * DMODEL * 2;
    _Float16* pwh   = (_Float16*)(ws + off); off += (size_t)DMODEL * DMODEL * 2;
    _Float16* attnh = (_Float16*)(ws + off); off += (size_t)M * DMODEL * 2;

    // 0) One-time fp32 -> f16 conversions of GEMM operands
    int nx = M * DMODEL / 4;
    f32_to_f16_kernel<<<(nx + 255) / 256, 256, 0, stream>>>(x, xh, nx);
    int nqw = 3 * DMODEL * DMODEL / 4;
    f32_to_f16_kernel<<<(nqw + 255) / 256, 256, 0, stream>>>(qkv_w, qwh, nqw);
    int npw = DMODEL * DMODEL / 4;
    f32_to_f16_kernel<<<(npw + 255) / 256, 256, 0, stream>>>(proj_w, pwh, npw);

    // 1) QKV projection: qkv[4096,3072] = xh * qwh^T + qkv_b
    dim3 g1(3 * DMODEL / 128, M / 256);
    gemm_wmma_kernel<<<g1, 256, 0, stream>>>(xh, qwh, qkv_b, qkv, M, 3 * DMODEL, DMODEL);

    // 2) RoPE + pack q,k,v to f16 [b,h,n,64]
    int total = BQ * NH * NSEQ * (HD / 2);
    rope_pack_kernel<<<total / 256, 256, 0, stream>>>(qkv, rope, qh, kh, vh);

    // 3) Flash attention -> attnh f16 [b,n,ch]
    flash_attn_kernel<<<BQ * NH * (NSEQ / 64), 128, 0, stream>>>(qh, kh, vh, attnh);

    // 4) Output projection: out = attnh * pwh^T + proj_b
    dim3 g4(DMODEL / 128, M / 256);
    gemm_wmma_kernel<<<g4, 256, 0, stream>>>(attnh, pwh, proj_b, out, M, DMODEL, DMODEL);
}
